// MATCHING_89412629168433
// MI455X (gfx1250) — matrix operand order repaired and checked by reading, not compiled, not run on hardware
//
#include <hip/hip_runtime.h>
#include <math.h>

#define EPSV 1e-8f

typedef __attribute__((ext_vector_type(16))) __bf16 v16bf;
typedef __attribute__((ext_vector_type(8)))  float  v8f;
typedef __attribute__((ext_vector_type(4)))  float  vf4;
typedef __attribute__((ext_vector_type(2)))  float  vf2;

#define B_   16
#define S_   512
#define H_   256
#define L_   16

#define TI 16
#define TJ 32

// LDS strides (padded to avoid bank conflicts on row-indexed lane reads)
#define V1_STRIDE 260      // f32 units: 4-bank skew per row
#define W2_STRIDE 256      // f32 units: broadcast reads, no pad needed
#define V2_STRIDE 264      // bf16 units: 4-bank skew per row

#define V1_BYTES (TI * V1_STRIDE * 4)                 // 16640
#define W2_BYTES (L_ * W2_STRIDE * 4)                 // 16384
#define V2_BYTES (TJ * V2_STRIDE * 2)                 // 16896
#define SMEM_BYTES (V1_BYTES + W2_BYTES + V2_BYTES)   // 49920 B (> 32 KB staging, union-safe)

// ---------------------------------------------------------------------------
// Pass 1: norms  nrm[b,s,l] = sqrt( sum_h (w[l,h]*v[b,s,h])^2 )
// One wave per (b,s) row; w^2 cached in LDS; shfl tree reduction (wave32).
// ---------------------------------------------------------------------------
__global__ void __launch_bounds__(256)
norm_kernel(const float* __restrict__ v, const float* __restrict__ w,
            float* __restrict__ nrm, int S) {
  __shared__ float w2s[L_ * H_];
  const int tid = threadIdx.x;
  for (int idx = tid; idx < L_ * H_; idx += 256) {
    float x = w[idx];
    w2s[idx] = x * x;
  }
  __syncthreads();

  const int lane = tid & 31;
  const int wave = tid >> 5;
  const int sIdx = blockIdx.x * 8 + wave;        // 0 .. B*S-1
  const int b = sIdx / S;
  const int s = sIdx - b * S;

  const float* vp = v + ((size_t)b * S + s) * H_;
  float vals2[8];
#pragma unroll
  for (int k = 0; k < 8; ++k) {
    float x = vp[lane + 32 * k];                 // coalesced
    vals2[k] = x * x;
  }

  float* np = nrm + ((size_t)b * S + s) * L_;
  for (int l = 0; l < L_; ++l) {
    const float* wr = w2s + l * H_;
    float acc = 0.f;
#pragma unroll
    for (int k = 0; k < 8; ++k) acc += wr[lane + 32 * k] * vals2[k];
#pragma unroll
    for (int off = 16; off >= 1; off >>= 1)
      acc += __shfl_down(acc, off, 32);
    if (lane == 0) np[l] = sqrtf(acc);
  }
}

// ---------------------------------------------------------------------------
// Pass 2: per (b, i-tile=16, j-tile=32) block: 16 per-l GEMM tiles via
// v_wmma_f32_16x16x32_bf16, normalize with v_rcp_f32, stage in LDS,
// coalesced writeback. Wave lw handles l in {2lw, 2lw+1}, both j sub-tiles.
// ---------------------------------------------------------------------------
__global__ void __launch_bounds__(256)
match_kernel(const float* __restrict__ v1, const float* __restrict__ v2,
             const float* __restrict__ w, const float* __restrict__ n1,
             const float* __restrict__ n2, float* __restrict__ out) {
  __shared__ __align__(16) char smem[SMEM_BYTES];
  float*  v1s  = (float*)smem;
  float*  w2s  = (float*)(smem + V1_BYTES);
  __bf16* v2s  = (__bf16*)(smem + V1_BYTES + W2_BYTES);
  float*  outS = (float*)smem;   // reused after compute phase

  const int tid  = threadIdx.x;
  const int lane = tid & 31;
  const int lw   = tid >> 5;                 // wave id -> l pair
  const int b    = blockIdx.z;
  const int i0   = blockIdx.y * TI;
  const int j0   = blockIdx.x * TJ;

  // ---- stage tiles into LDS ----
  const float* v1g = v1 + (((size_t)b * S_) + i0) * H_;
#pragma unroll
  for (int p = 0; p < 4; ++p) {              // 16x256 f32, float4 loads
    int f = tid + p * 256;
    int row = f >> 6, c4 = (f & 63) << 2;
    vf4 d = *(const vf4*)(v1g + row * H_ + c4);
    *(vf4*)(v1s + row * V1_STRIDE + c4) = d;
  }
#pragma unroll
  for (int p = 0; p < 4; ++p) {              // w^2: 16x256 f32
    int f = tid + p * 256;
    vf4 d = *(const vf4*)(w + f * 4);
    d = d * d;
    *(vf4*)(w2s + f * 4) = d;
  }
  const float* v2g = v2 + (((size_t)b * S_) + j0) * H_;
#pragma unroll
  for (int p = 0; p < 8; ++p) {              // 32x256 f32 -> bf16
    int f = tid + p * 256;
    int row = f >> 6, c4 = (f & 63) << 2;
    vf4 d = *(const vf4*)(v2g + row * H_ + c4);
    __bf16* dst = v2s + row * V2_STRIDE + c4;
    dst[0] = (__bf16)d.x; dst[1] = (__bf16)d.y;
    dst[2] = (__bf16)d.z; dst[3] = (__bf16)d.w;
  }
  __syncthreads();

  // ---- WMMA main loop ----
  v8f acc[2][2];
#pragma unroll
  for (int li = 0; li < 2; ++li)
#pragma unroll
    for (int jt = 0; jt < 2; ++jt)
      acc[li][jt] = (v8f){0.f, 0.f, 0.f, 0.f, 0.f, 0.f, 0.f, 0.f};

  const int m   = lane & 15;                 // A row / B column
  const int kb  = (lane >> 4) << 3;          // A: +0 / +8 K skew per lane half
  const int kbB = (lane >> 4) << 3;          // B: +0 / +8 K skew per lane half
  const float* v1row  = v1s + m * V1_STRIDE;
  const float* w2row0 = w2s + (2 * lw) * W2_STRIDE;
  const float* w2row1 = w2row0 + W2_STRIDE;

  for (int k0 = 0; k0 < H_; k0 += 32) {
    // B fragments (dense bf16, K x N: same K halves of 8 as A)
    v16bf bf[2];
#pragma unroll
    for (int jt = 0; jt < 2; ++jt) {
      const __bf16* br = v2s + (jt * 16 + m) * V2_STRIDE + k0 + kbB;
#pragma unroll
      for (int e = 0; e < 16; ++e) bf[jt][e] = br[(e & 7) + ((e >> 3) << 4)];
    }
    // A fragments: bf16( w^2[l,k] * v1[i,k] ), 16-bit A layout (K halves of 8)
    v16bf af[2];
#pragma unroll
    for (int li = 0; li < 2; ++li) {
      const float* wr = li ? w2row1 : w2row0;
#pragma unroll
      for (int h = 0; h < 2; ++h) {
        const float* vp = v1row + k0 + h * 16 + kb;
        const float* wp = wr    + k0 + h * 16 + kb;
#pragma unroll
        for (int e = 0; e < 8; ++e)
          af[li][h * 8 + e] = (__bf16)(vp[e] * wp[e]);
      }
    }
#pragma unroll
    for (int li = 0; li < 2; ++li)
#pragma unroll
      for (int jt = 0; jt < 2; ++jt)
        acc[li][jt] = __builtin_amdgcn_wmma_f32_16x16x32_bf16(
            false, af[li], false, bf[jt], (short)0, acc[li][jt], false, false);
  }
  __syncthreads();   // all waves done reading compute LDS

  // ---- normalize (fast v_rcp_f32) + stage result tile (l-major) in LDS ----
  const int Mb = (lane >> 4) << 3;           // C/D: M = r + 8*(lane>=16)
  const int N  = lane & 15;
  const int l0 = 2 * lw;
  const float* n1p = n1 + (((size_t)b * S_) + i0) * L_;
  const float* n2p = n2 + (((size_t)b * S_) + j0) * L_;

  float a0[8], a1[8];                        // n1 rows, hoisted across jt
#pragma unroll
  for (int r = 0; r < 8; ++r) {
    a0[r] = n1p[(Mb + r) * L_ + l0];
    a1[r] = n1p[(Mb + r) * L_ + l0 + 1];
  }
#pragma unroll
  for (int jt = 0; jt < 2; ++jt) {
    const float nn20 = n2p[(jt * 16 + N) * L_ + l0];
    const float nn21 = n2p[(jt * 16 + N) * L_ + l0 + 1];
#pragma unroll
    for (int r = 0; r < 8; ++r) {
      float r0 = __builtin_amdgcn_rcpf(a0[r] * nn20 + EPSV);
      float r1 = __builtin_amdgcn_rcpf(a1[r] * nn21 + EPSV);
      vf2 val;
      val.x = acc[0][jt][r] * r0;
      val.y = acc[1][jt][r] * r1;
      *(vf2*)(outS + (Mb + r) * (TJ * L_) + (jt * 16 + N) * L_ + l0) = val;
    }
  }
  __syncthreads();

  // ---- coalesced writeback: 16 rows x 512 contiguous floats each ----
  float* og = out + ((((size_t)b * S_) + i0) * S_ + j0) * L_;
#pragma unroll
  for (int p = 0; p < 8; ++p) {
    int f = tid + p * 256;                   // 2048 float4 total
    int row = f >> 7, c4 = (f & 127) << 2;
    vf4 d = *(const vf4*)(outS + row * (TJ * L_) + c4);
    *(vf4*)(og + (size_t)row * (S_ * L_) + c4) = d;
  }
}

// ---------------------------------------------------------------------------
extern "C" void kernel_launch(void* const* d_in, const int* in_sizes, int n_in,
                              void* d_out, int out_size, void* d_ws, size_t ws_size,
                              hipStream_t stream) {
  const float* v1 = (const float*)d_in[0];
  const float* v2 = (const float*)d_in[1];
  const float* w  = (const float*)d_in[2];
  float* out = (float*)d_out;

  float* n1 = (float*)d_ws;                          // B*S*L floats
  float* n2 = n1 + (size_t)B_ * S_ * L_;             // B*S*L floats

  norm_kernel<<<dim3(B_ * S_ / 8), 256, 0, stream>>>(v1, w, n1, S_);
  norm_kernel<<<dim3(B_ * S_ / 8), 256, 0, stream>>>(v2, w, n2, S_);
  match_kernel<<<dim3(S_ / TJ, S_ / TI, B_), 256, 0, stream>>>(v1, v2, w, n1, n2, out);
}
